// GINEncoder_46540265619882
// MI455X (gfx1250) — matrix operand order repaired and checked by reading, not compiled, not run on hardware
//
#include <hip/hip_runtime.h>

// ---------------- problem constants (fixed by the reference) ----------------
#define NNODES 46080
#define NEDGES 1048576
#define BSZ    128
#define NPG    360           // nodes per graph
#define FDIM   360
#define HDIM   256
#define ODIM   512
#define EMBD   128
#define CIN    616           // FDIM + HDIM
#define CINP   640           // CIN padded to multiple of 32 for WMMA K

// ---------------- types ----------------
typedef __attribute__((ext_vector_type(16))) __bf16        v16bf;
typedef __attribute__((ext_vector_type(8)))  float         v8f;
typedef __attribute__((ext_vector_type(4)))  unsigned int  u32x4;

union Frag { v16bf v; u32x4 q[2]; };

__device__ __forceinline__ unsigned short f2bf(float f) {
  unsigned u = __float_as_uint(f);
  u += 0x7FFFu + ((u >> 16) & 1u);     // round-to-nearest-even
  return (unsigned short)(u >> 16);
}

// ---------------- weight convert + transpose: src f32 [K][Nout] -> dst bf16 [Nout][Kpad] ----
__global__ void k_wcvt(const float* __restrict__ src, unsigned short* __restrict__ dst,
                       int K, int Nout, int Kpad) {
  int t = blockIdx.x * blockDim.x + threadIdx.x;
  int total = Nout * Kpad;
  if (t >= total) return;
  int n = t / Kpad, k = t - n * Kpad;
  dst[t] = (k < K) ? f2bf(src[(size_t)k * Nout + n]) : (unsigned short)0;
}

// ---------------- h0 = concat(x, tile(emb_table)) : fp32 [N][616] ----------------
__global__ void k_concat(const float* __restrict__ x, const float* __restrict__ emb,
                         float* __restrict__ h0) {
  int i = blockIdx.x;
  int loc = i % NPG;
  float* out = h0 + (size_t)i * CIN;
  const float* xi = x  + (size_t)i * FDIM;
  const float* ei = emb + (size_t)loc * HDIM;
  for (int f = threadIdx.x; f < CIN; f += blockDim.x)
    out[f] = (f < FDIM) ? xi[f] : ei[f - FDIM];
}

// ---------------- zero n floats (n multiple of 4) ----------------
__global__ void k_zero(float* __restrict__ p, long n) {
  long t = ((long)blockIdx.x * blockDim.x + threadIdx.x) * 4;
  long stride = (long)gridDim.x * blockDim.x * 4;
  for (long i = t; i < n; i += stride)
    *(float4*)(p + i) = make_float4(0.f, 0.f, 0.f, 0.f);
}

// ---------------- edge scatter: agg[dst] += h[src], fp32 atomics (D mult of 4) ----------------
__global__ void k_scatter(const float* __restrict__ h, const int* __restrict__ ei,
                          float* __restrict__ agg, int D) {
  int e = blockIdx.x;
  int src = ei[e];
  int dst = ei[NEDGES + e];
  const float* hs = h + (size_t)src * D;
  float* ad = agg + (size_t)dst * D;
  for (int c = threadIdx.x * 4; c < D; c += blockDim.x * 4) {
    float4 v = *(const float4*)(hs + c);
    atomicAdd(ad + c + 0, v.x);
    atomicAdd(ad + c + 1, v.y);
    atomicAdd(ad + c + 2, v.z);
    atomicAdd(ad + c + 3, v.w);
  }
}

// ---------------- zb = bf16(agg + h), K-padded with zeros ----------------
__global__ void k_addcvt(const float* __restrict__ agg, const float* __restrict__ h,
                         unsigned short* __restrict__ zb, int D, int Kpad) {
  int i = blockIdx.x;
  const float* a  = agg + (size_t)i * D;
  const float* hh = h   + (size_t)i * D;
  unsigned short* z = zb + (size_t)i * Kpad;
  for (int k = threadIdx.x; k < Kpad; k += blockDim.x)
    z[k] = (k < D) ? f2bf(a[k] + hh[k]) : (unsigned short)0;
}

// ---------------- bf16 WMMA GEMM: out[M][Nout] = relu?(A[M][K] * W^T + bias) ----------------
// A row-major bf16 (lda=Kpad), Bt = weights transposed bf16 [Nout][Kpad] (ldb=Kpad).
// One wave (32 threads) computes a 16x64 output tile. grid = (M/16, Nout/64).
template <int RELU, int OUT_BF16, int OUT_F32>
__global__ void k_gemm(const unsigned short* __restrict__ A, int lda,
                       const unsigned short* __restrict__ Bt, int ldb,
                       const float* __restrict__ bias,
                       unsigned short* __restrict__ Obf,
                       float* __restrict__ Of,
                       int ldo, int K) {
  const int lane = threadIdx.x;        // wave32
  const int mt   = blockIdx.x;
  const int n0   = blockIdx.y << 6;
  const int nn   = lane & 15;
  const int kg   = lane >> 4;          // 0 or 1

  const unsigned short* Ap  = A  + (size_t)((mt << 4) + nn) * lda + kg * 16;
  const unsigned short* Bp0 = Bt + (size_t)(n0 + nn +  0) * ldb + kg * 16;
  const unsigned short* Bp1 = Bt + (size_t)(n0 + nn + 16) * ldb + kg * 16;
  const unsigned short* Bp2 = Bt + (size_t)(n0 + nn + 32) * ldb + kg * 16;
  const unsigned short* Bp3 = Bt + (size_t)(n0 + nn + 48) * ldb + kg * 16;

  v8f acc0 = {}, acc1 = {}, acc2 = {}, acc3 = {};

  for (int k = 0; k < K; k += 32) {
    Frag a, b0, b1, b2, b3;
    a.q[0]  = *(const u32x4*)(Ap  + k);
    a.q[1]  = *(const u32x4*)(Ap  + k + 8);
    b0.q[0] = *(const u32x4*)(Bp0 + k);
    b0.q[1] = *(const u32x4*)(Bp0 + k + 8);
    b1.q[0] = *(const u32x4*)(Bp1 + k);
    b1.q[1] = *(const u32x4*)(Bp1 + k + 8);
    b2.q[0] = *(const u32x4*)(Bp2 + k);
    b2.q[1] = *(const u32x4*)(Bp2 + k + 8);
    b3.q[0] = *(const u32x4*)(Bp3 + k);
    b3.q[1] = *(const u32x4*)(Bp3 + k + 8);
    acc0 = __builtin_amdgcn_wmma_f32_16x16x32_bf16(false, a.v, false, b0.v, (short)0, acc0, false, false);
    acc1 = __builtin_amdgcn_wmma_f32_16x16x32_bf16(false, a.v, false, b1.v, (short)0, acc1, false, false);
    acc2 = __builtin_amdgcn_wmma_f32_16x16x32_bf16(false, a.v, false, b2.v, (short)0, acc2, false, false);
    acc3 = __builtin_amdgcn_wmma_f32_16x16x32_bf16(false, a.v, false, b3.v, (short)0, acc3, false, false);
  }

  const float bv0 = bias[n0 + nn +  0];
  const float bv1 = bias[n0 + nn + 16];
  const float bv2 = bias[n0 + nn + 32];
  const float bv3 = bias[n0 + nn + 48];

#pragma unroll
  for (int r = 0; r < 8; ++r) {
    const size_t row = (size_t)(mt << 4) + kg * 8 + r;
    float v0 = acc0[r] + bv0;
    float v1 = acc1[r] + bv1;
    float v2 = acc2[r] + bv2;
    float v3 = acc3[r] + bv3;
    if (RELU) { v0 = fmaxf(v0, 0.f); v1 = fmaxf(v1, 0.f); v2 = fmaxf(v2, 0.f); v3 = fmaxf(v3, 0.f); }
    if (OUT_F32) {
      Of[row * ldo + n0 + nn +  0] = v0;
      Of[row * ldo + n0 + nn + 16] = v1;
      Of[row * ldo + n0 + nn + 32] = v2;
      Of[row * ldo + n0 + nn + 48] = v3;
    }
    if (OUT_BF16) {
      Obf[row * ldo + n0 + nn +  0] = f2bf(v0);
      Obf[row * ldo + n0 + nn + 16] = f2bf(v1);
      Obf[row * ldo + n0 + nn + 32] = f2bf(v2);
      Obf[row * ldo + n0 + nn + 48] = f2bf(v3);
    }
  }
}

// ---------------- mean pooling over 360 nodes per graph: g[B][512] ----------------
__global__ void k_pool(const float* __restrict__ h, float* __restrict__ g) {
  int t = blockIdx.x * blockDim.x + threadIdx.x;   // b*512 + c
  if (t >= BSZ * ODIM) return;
  int b = t >> 9, c = t & 511;
  const float* p = h + (size_t)b * NPG * ODIM + c;
  float s = 0.f;
  for (int nn = 0; nn < NPG; ++nn) s += p[(size_t)nn * ODIM];
  g[t] = s * (1.0f / (float)NPG);
}

// ---------------- final projection: out[B][128] = g[B][512] @ w_out + b_out (fp32) ----------
__global__ void k_final(const float* __restrict__ g, const float* __restrict__ w,
                        const float* __restrict__ bias, float* __restrict__ out) {
  int t = blockIdx.x * blockDim.x + threadIdx.x;   // b*128 + e
  if (t >= BSZ * EMBD) return;
  int b = t >> 7, e = t & 127;
  const float* gb = g + (size_t)b * ODIM;
  float s = bias[e];
  for (int c = 0; c < ODIM; ++c) s += gb[c] * w[(size_t)c * EMBD + e];
  out[t] = s;
}

// ============================== launcher ==============================
extern "C" void kernel_launch(void* const* d_in, const int* in_sizes, int n_in,
                              void* d_out, int out_size, void* d_ws, size_t ws_size,
                              hipStream_t stream) {
  const float* x        = (const float*)d_in[0];
  const int*   ei       = (const int*)  d_in[1];
  const float* embt     = (const float*)d_in[3];
  const float* w1a = (const float*)d_in[4],  *b1a = (const float*)d_in[5];
  const float* w1b = (const float*)d_in[6],  *b1b = (const float*)d_in[7];
  const float* w2a = (const float*)d_in[8],  *b2a = (const float*)d_in[9];
  const float* w2b = (const float*)d_in[10], *b2b = (const float*)d_in[11];
  const float* w3a = (const float*)d_in[12], *b3a = (const float*)d_in[13];
  const float* w3b = (const float*)d_in[14], *b3b = (const float*)d_in[15];
  const float* wo  = (const float*)d_in[16], *bo  = (const float*)d_in[17];
  float* out = (float*)d_out;

  char* ws = (char*)d_ws;
  auto al = [](size_t v) { return (v + 255) & ~(size_t)255; };
  size_t off = 0;
  auto take = [&](size_t bytes) { size_t o = off; off += al(bytes); return o; };

  float*          h0   = (float*)         (ws + take((size_t)NNODES * CIN * 4));   // also conv3 out (N x 512 fits)
  float*          agg  = (float*)         (ws + take((size_t)NNODES * CIN * 4));
  unsigned short* zb   = (unsigned short*)(ws + take((size_t)NNODES * CINP * 2));
  unsigned short* y1   = (unsigned short*)(ws + take((size_t)NNODES * ODIM * 2));
  float*          hA   = (float*)         (ws + take((size_t)NNODES * HDIM * 4));
  float*          hB   = (float*)         (ws + take((size_t)NNODES * HDIM * 4));
  unsigned short* w1at = (unsigned short*)(ws + take((size_t)HDIM * CINP * 2));
  unsigned short* w1bt = (unsigned short*)(ws + take((size_t)HDIM * HDIM * 2));
  unsigned short* w2at = (unsigned short*)(ws + take((size_t)HDIM * HDIM * 2));
  unsigned short* w2bt = (unsigned short*)(ws + take((size_t)HDIM * HDIM * 2));
  unsigned short* w3at = (unsigned short*)(ws + take((size_t)ODIM * HDIM * 2));
  unsigned short* w3bt = (unsigned short*)(ws + take((size_t)ODIM * ODIM * 2));
  float*          g    = (float*)         (ws + take((size_t)BSZ * ODIM * 4));
  float*          hC   = h0;  // conv3 output aliases h0 (dead after conv1)

  // --- weights: f32 [K][Nout] -> bf16 transposed [Nout][Kpad] ---
  auto wc = [&](const float* s, unsigned short* d, int K, int Nout, int Kpad) {
    int total = Nout * Kpad;
    k_wcvt<<<(total + 255) / 256, 256, 0, stream>>>(s, d, K, Nout, Kpad);
  };
  wc(w1a, w1at, CIN,  HDIM, CINP);
  wc(w1b, w1bt, HDIM, HDIM, HDIM);
  wc(w2a, w2at, HDIM, HDIM, HDIM);
  wc(w2b, w2bt, HDIM, HDIM, HDIM);
  wc(w3a, w3at, HDIM, ODIM, HDIM);
  wc(w3b, w3bt, ODIM, ODIM, ODIM);

  // --- h0 = concat(x, emb) ---
  k_concat<<<NNODES, 256, 0, stream>>>(x, embt, h0);

  const dim3 wmma_blk(32, 1, 1);

  // ================= conv1: 616 -> 256 -> 256 =================
  {
    long n = (long)NNODES * CIN;
    k_zero<<<(unsigned)((n / 4 + 255) / 256), 256, 0, stream>>>(agg, n);
    k_scatter<<<NEDGES, 160, 0, stream>>>(h0, ei, agg, CIN);
    k_addcvt<<<NNODES, 256, 0, stream>>>(agg, h0, zb, CIN, CINP);
    k_gemm<1, 1, 0><<<dim3(NNODES / 16, HDIM / 64), wmma_blk, 0, stream>>>(
        zb, CINP, w1at, CINP, b1a, y1, nullptr, HDIM, CINP);
    k_gemm<1, 0, 1><<<dim3(NNODES / 16, HDIM / 64), wmma_blk, 0, stream>>>(
        y1, HDIM, w1bt, HDIM, b1b, nullptr, hA, HDIM, HDIM);
  }

  // ================= conv2: 256 -> 256 -> 256 =================
  {
    long n = (long)NNODES * HDIM;
    k_zero<<<(unsigned)((n / 4 + 255) / 256), 256, 0, stream>>>(agg, n);
    k_scatter<<<NEDGES, 64, 0, stream>>>(hA, ei, agg, HDIM);
    k_addcvt<<<NNODES, 256, 0, stream>>>(agg, hA, zb, HDIM, HDIM);
    k_gemm<1, 1, 0><<<dim3(NNODES / 16, HDIM / 64), wmma_blk, 0, stream>>>(
        zb, HDIM, w2at, HDIM, b2a, y1, nullptr, HDIM, HDIM);
    k_gemm<1, 0, 1><<<dim3(NNODES / 16, HDIM / 64), wmma_blk, 0, stream>>>(
        y1, HDIM, w2bt, HDIM, b2b, nullptr, hB, HDIM, HDIM);
  }

  // ================= conv3: 256 -> 512 -> 512 (no outer relu) =================
  {
    long n = (long)NNODES * HDIM;
    k_zero<<<(unsigned)((n / 4 + 255) / 256), 256, 0, stream>>>(agg, n);
    k_scatter<<<NEDGES, 64, 0, stream>>>(hB, ei, agg, HDIM);
    k_addcvt<<<NNODES, 256, 0, stream>>>(agg, hB, zb, HDIM, HDIM);
    k_gemm<1, 1, 0><<<dim3(NNODES / 16, ODIM / 64), wmma_blk, 0, stream>>>(
        zb, HDIM, w3at, HDIM, b3a, y1, nullptr, ODIM, HDIM);
    k_gemm<0, 0, 1><<<dim3(NNODES / 16, ODIM / 64), wmma_blk, 0, stream>>>(
        y1, ODIM, w3bt, ODIM, b3b, nullptr, hC, ODIM, ODIM);
  }

  // ================= pooling + final projection =================
  k_pool<<<(BSZ * ODIM + 255) / 256, 256, 0, stream>>>(hC, g);
  k_final<<<(BSZ * EMBD + 255) / 256, 256, 0, stream>>>(g, wo, bo, out);
}
